// BiLSTMTagger_4612794876775
// MI455X (gfx1250) — matrix operand order repaired and checked by reading, not compiled, not run on hardware
//
#include <hip/hip_runtime.h>
#include <hip/hip_bf16.h>
#include <math.h>

#define SEQ 512
#define HD  200          // hidden per direction
#define HD2 100          // HD/2 (packed bf16 pairs)
#define G4  800          // 4*HD gates
#define INF_ 400         // feature width
#define KP  416          // K padded to multiple of 32 (13*32)

typedef __attribute__((ext_vector_type(16))) __bf16 v16bf;
typedef __attribute__((ext_vector_type(8)))  __bf16 v8bf;
typedef __attribute__((ext_vector_type(8)))  float  v8f;

// ---------- prep kernels ----------

__global__ void k_convert_pad(const float* __restrict__ src, __bf16* __restrict__ dst,
                              int R, int C, int Cpad) {
  int idx = blockIdx.x * blockDim.x + threadIdx.x;
  if (idx >= R * Cpad) return;
  int r = idx / Cpad, c = idx - r * Cpad;
  float v = (c < C) ? src[(size_t)r * C + c] : 0.f;
  dst[idx] = (__bf16)v;
}

__global__ void k_transpose_convert_pad(const float* __restrict__ src, __bf16* __restrict__ dst,
                                        int D, int E, int Dpad) {
  // dst[e*Dpad + d] = src[d*E + e]  (pad with 0)
  int idx = blockIdx.x * blockDim.x + threadIdx.x;
  if (idx >= E * Dpad) return;
  int e = idx / Dpad, d = idx - e * Dpad;
  dst[idx] = (__bf16)((d < D) ? src[(size_t)d * E + e] : 0.f);
}

// Whh [800,200] fp32 -> packed bf16 pairs, k-major: out[k2*800 + r] = {W[r][2k2], W[r][2k2+1]}
__global__ void k_pack_whh(const float* __restrict__ Whh, unsigned int* __restrict__ out) {
  int idx = blockIdx.x * blockDim.x + threadIdx.x;
  if (idx >= HD2 * G4) return;
  int k2 = idx / G4, r = idx - k2 * G4;
  unsigned int lo = __builtin_bit_cast(unsigned short, (__bf16)Whh[(size_t)r * HD + 2 * k2]);
  unsigned int hi = __builtin_bit_cast(unsigned short, (__bf16)Whh[(size_t)r * HD + 2 * k2 + 1]);
  out[idx] = lo | (hi << 16);
}

__global__ void k_add(const float* __restrict__ a, const float* __restrict__ b,
                      float* __restrict__ o, int n) {
  int idx = blockIdx.x * blockDim.x + threadIdx.x;
  if (idx < n) o[idx] = a[idx] + b[idx];
}

__global__ void k_embed_bf(const float* __restrict__ wemb, const float* __restrict__ pemb,
                           const int* __restrict__ widx, const int* __restrict__ pidx,
                           __bf16* __restrict__ x) {
  int idx = blockIdx.x * blockDim.x + threadIdx.x;
  if (idx >= SEQ * KP) return;
  int t = idx / KP, j = idx - t * KP;
  float v = 0.f;
  if (j < 300)       v = wemb[(size_t)widx[t] * 300 + j];
  else if (j < INF_) v = pemb[(size_t)pidx[t] * 100 + (j - 300)];
  x[idx] = (__bf16)v;
}

// zero the K-pad columns (400..415) of the 7 bf16 activation buffers written partially later
__global__ void k_zero_pads(__bf16* b0, __bf16* b1, __bf16* b2, __bf16* b3,
                            __bf16* b4, __bf16* b5, __bf16* b6) {
  const int per = SEQ * (KP - INF_);   // 512*16
  int gid = blockIdx.x * blockDim.x + threadIdx.x;
  if (gid >= 7 * per) return;
  int b = gid / per, rem = gid - b * per;
  int row = rem >> 4, cc = rem & 15;
  __bf16* p = (b == 0) ? b0 : (b == 1) ? b1 : (b == 2) ? b2 : (b == 3) ? b3
            : (b == 4) ? b4 : (b == 5) ? b5 : b6;
  p[(size_t)row * KP + INF_ + cc] = (__bf16)0.f;
}

// ---------- WMMA bf16 GEMM: C[M,N] = A[M,Kp] * B[N,Kp]^T (+bias)(+relu) ----------
// One wave computes a 64x16 strip of C (4 WMMA tiles along M share one B fragment).
// Optionally writes fp32 C and/or a KP-padded bf16 copy for the next stage.

__global__ void k_gemm_bf16_wmma(const __bf16* __restrict__ A,
                                 const __bf16* __restrict__ B,
                                 float* __restrict__ C,        // nullable
                                 __bf16* __restrict__ Cbf,     // nullable, row stride KP
                                 int M, int N, int Kpad,
                                 const float* __restrict__ bias_vec,
                                 const float* __restrict__ bias_scalar,
                                 int relu) {
  const int lane = threadIdx.x & 31;
  const int half = lane >> 4;    // 0 or 1
  const int l16  = lane & 15;
  const int tn = blockIdx.x, tm = blockIdx.y;

  const __bf16* brow = B + (size_t)(tn * 16 + l16) * Kpad;   // C column for this lane
  const __bf16* arow[4];
#pragma unroll
  for (int s = 0; s < 4; ++s)
    arow[s] = A + (size_t)(tm * 64 + s * 16 + l16) * Kpad;

  v8f acc[4] = {};
  for (int k0 = 0; k0 < Kpad; k0 += 32) {
    // B fragment (32x16): lane half 0 -> K {0..15}; half 1 -> K {16..31}, contiguous
    v8bf b0 = *(const v8bf*)(brow + k0 + half * 16);
    v8bf b1 = *(const v8bf*)(brow + k0 + half * 16 + 8);
    v16bf bv = __builtin_shufflevector(b0, b1, 0,1,2,3,4,5,6,7,8,9,10,11,12,13,14,15);
#pragma unroll
    for (int s = 0; s < 4; ++s) {
      // A fragment (16x32): lane half 0 -> K {0..15}; half 1 -> K {16..31}, contiguous
      v8bf a0 = *(const v8bf*)(arow[s] + k0 + half * 16);
      v8bf a1 = *(const v8bf*)(arow[s] + k0 + half * 16 + 8);
      v16bf av = __builtin_shufflevector(a0, a1, 0,1,2,3,4,5,6,7,8,9,10,11,12,13,14,15);
      acc[s] = __builtin_amdgcn_wmma_f32_16x16x32_bf16(
          /*neg_a=*/false, av, /*neg_b=*/false, bv,
          /*c_mod=*/(short)0, acc[s], /*reuse_a=*/false, /*reuse_b=*/false);
    }
  }

  const int col = tn * 16 + l16;
  float bb = 0.f;
  if (bias_vec)    bb  = bias_vec[col];
  if (bias_scalar) bb += bias_scalar[0];
#pragma unroll
  for (int s = 0; s < 4; ++s) {
#pragma unroll
    for (int r = 0; r < 8; ++r) {
      int row = tm * 64 + s * 16 + r + half * 8;
      float v = acc[s][r] + bb;
      if (relu && v < 0.f) v = 0.f;
      if (C)   C[(size_t)row * N + col]  = v;
      if (Cbf) Cbf[(size_t)row * KP + col] = (__bf16)v;
    }
  }
}

// ---------- sequential bidirectional LSTM scan ----------
// One 800-thread block per direction. Whh (packed bf16 pairs, k-major [100,800]) is
// staged once into 320,000 B of LDS (CDNA5: 320KB LDS per WGP); h/c recurrence in fp32.
// pre*: [SEQ,800] = x@Wih.T + (bih+bhh). Output written directly as bf16 into x_bf[SEQ,KP].

__global__ void __launch_bounds__(G4)
k_lstm_bidir(const float* __restrict__ preF, const float* __restrict__ preB,
             const unsigned int* __restrict__ WpkF, const unsigned int* __restrict__ WpkB,
             __bf16* __restrict__ xout_bf) {
  __shared__ unsigned int Wlds[HD2 * G4];             // 80,000 u32 = 320,000 B
  __shared__ __align__(16) float h_sh[HD];
  __shared__ float g_sh[G4];
  const int dir = blockIdx.x;
  const float* __restrict__ pre = dir ? preB : preF;
  const unsigned int* __restrict__ Wg = dir ? WpkB : WpkF;
  const int r = threadIdx.x;  // 0..799

  for (int i = r; i < HD2 * G4; i += G4) Wlds[i] = Wg[i];   // stage weights into LDS
  float c = 0.f;
  if (r < HD) h_sh[r] = 0.f;
  __syncthreads();

  const float2* __restrict__ h2 = (const float2*)h_sh;
  for (int step = 0; step < SEQ; ++step) {
    const int t = dir ? (SEQ - 1 - step) : step;
    float acc = pre[(size_t)t * G4 + r];
#pragma unroll 4
    for (int k2 = 0; k2 < HD2; ++k2) {
      unsigned int w = Wlds[k2 * G4 + r];   // coalesced across lanes
      float2 hp = h2[k2];                    // broadcast
      float wlo = (float)__builtin_bit_cast(__bf16, (unsigned short)(w & 0xffffu));
      float whi = (float)__builtin_bit_cast(__bf16, (unsigned short)(w >> 16));
      acc = fmaf(wlo, hp.x, acc);
      acc = fmaf(whi, hp.y, acc);
    }
    g_sh[r] = acc;
    __syncthreads();
    if (r < HD) {
      float ig = 1.f / (1.f + __expf(-g_sh[r]));
      float fg = 1.f / (1.f + __expf(-g_sh[HD + r]));
      float gg = tanhf(g_sh[2 * HD + r]);
      float og = 1.f / (1.f + __expf(-g_sh[3 * HD + r]));
      c = fg * c + ig * gg;
      float h = og * tanhf(c);
      h_sh[r] = h;
      xout_bf[(size_t)t * KP + dir * HD + r] = (__bf16)h;
    }
    __syncthreads();
  }
}

// ---------- host orchestration ----------

static inline int ceil_div(int a, int b) { return (a + b - 1) / b; }

extern "C" void kernel_launch(void* const* d_in, const int* in_sizes, int n_in,
                              void* d_out, int out_size, void* d_ws, size_t ws_size,
                              hipStream_t stream) {
  const float* word_emb = (const float*)d_in[0];
  const float* pos_emb  = (const float*)d_in[1];
  const float* Wih      = (const float*)d_in[2];   // [2,2,800,400]
  const float* Whh      = (const float*)d_in[3];   // [2,2,800,200]
  const float* bih      = (const float*)d_in[4];   // [2,2,800]
  const float* bhh      = (const float*)d_in[5];
  const float* W_h1 = (const float*)d_in[6];  const float* b_h1 = (const float*)d_in[7];
  const float* W_h2 = (const float*)d_in[8];  const float* b_h2 = (const float*)d_in[9];
  const float* W_d1 = (const float*)d_in[10]; const float* b_d1 = (const float*)d_in[11];
  const float* W_d2 = (const float*)d_in[12]; const float* b_d2 = (const float*)d_in[13];
  const float* W_bi = (const float*)d_in[14]; const float* b_bi = (const float*)d_in[15];
  const int* widx = (const int*)d_in[16];
  const int* pidx = (const int*)d_in[17];
  float* out = (float*)d_out;   // [512,512]

  size_t off = 0;
  char* base = (char*)d_ws;
  auto alloc = [&](size_t bytes) -> void* {
    off = (off + 255) & ~(size_t)255;
    void* p = base + off;
    off += bytes;
    return p;
  };

  __bf16* wih_bf[4];     for (int i = 0; i < 4; ++i) wih_bf[i] = (__bf16*)alloc((size_t)G4 * KP * 2);
  unsigned int* wpk[4];  for (int i = 0; i < 4; ++i) wpk[i] = (unsigned int*)alloc((size_t)HD2 * G4 * 4);
  float*  bias4  = (float*)alloc(4 * G4 * 4);
  __bf16* wh1_bf = (__bf16*)alloc((size_t)INF_ * KP * 2);
  __bf16* wh2_bf = (__bf16*)alloc((size_t)INF_ * KP * 2);
  __bf16* wd1_bf = (__bf16*)alloc((size_t)INF_ * KP * 2);
  __bf16* wd2_bf = (__bf16*)alloc((size_t)INF_ * KP * 2);
  __bf16* wbiT_bf = (__bf16*)alloc((size_t)INF_ * KP * 2);
  __bf16* x0_bf  = (__bf16*)alloc((size_t)SEQ * KP * 2);
  float*  preF0  = (float*)alloc((size_t)SEQ * G4 * 4);
  float*  preB0  = (float*)alloc((size_t)SEQ * G4 * 4);
  float*  preF1  = (float*)alloc((size_t)SEQ * G4 * 4);
  float*  preB1  = (float*)alloc((size_t)SEQ * G4 * 4);
  __bf16* x1_bf  = (__bf16*)alloc((size_t)SEQ * KP * 2);
  __bf16* x2_bf  = (__bf16*)alloc((size_t)SEQ * KP * 2);
  __bf16* h1_bf  = (__bf16*)alloc((size_t)SEQ * KP * 2);
  __bf16* head_bf = (__bf16*)alloc((size_t)SEQ * KP * 2);
  __bf16* d1_bf  = (__bf16*)alloc((size_t)SEQ * KP * 2);
  __bf16* dep_bf = (__bf16*)alloc((size_t)SEQ * KP * 2);
  __bf16* tmp_bf = (__bf16*)alloc((size_t)SEQ * KP * 2);
  (void)ws_size;

  const int TB = 256;
  for (int s = 0; s < 4; ++s) {
    k_convert_pad<<<ceil_div(G4 * KP, TB), TB, 0, stream>>>(
        Wih + (size_t)s * G4 * INF_, wih_bf[s], G4, INF_, KP);
    k_pack_whh<<<ceil_div(HD2 * G4, TB), TB, 0, stream>>>(
        Whh + (size_t)s * G4 * HD, wpk[s]);
  }
  k_add<<<ceil_div(4 * G4, TB), TB, 0, stream>>>(bih, bhh, bias4, 4 * G4);
  k_convert_pad<<<ceil_div(INF_ * KP, TB), TB, 0, stream>>>(W_h1, wh1_bf, INF_, INF_, KP);
  k_convert_pad<<<ceil_div(INF_ * KP, TB), TB, 0, stream>>>(W_h2, wh2_bf, INF_, INF_, KP);
  k_convert_pad<<<ceil_div(INF_ * KP, TB), TB, 0, stream>>>(W_d1, wd1_bf, INF_, INF_, KP);
  k_convert_pad<<<ceil_div(INF_ * KP, TB), TB, 0, stream>>>(W_d2, wd2_bf, INF_, INF_, KP);
  k_transpose_convert_pad<<<ceil_div(INF_ * KP, TB), TB, 0, stream>>>(W_bi, wbiT_bf, INF_, INF_, KP);

  k_embed_bf<<<ceil_div(SEQ * KP, TB), TB, 0, stream>>>(word_emb, pos_emb, widx, pidx, x0_bf);
  k_zero_pads<<<ceil_div(7 * SEQ * (KP - INF_), TB), TB, 0, stream>>>(
      x1_bf, x2_bf, h1_bf, head_bf, d1_bf, dep_bf, tmp_bf);

  dim3 blk32(32);
  dim3 grid_pre(G4 / 16, SEQ / 64);     // 50 x 8
  dim3 grid_400(INF_ / 16, SEQ / 64);   // 25 x 8
  dim3 grid_out(SEQ / 16, SEQ / 64);    // 32 x 8

  // ---- LSTM layer 0 ----
  k_gemm_bf16_wmma<<<grid_pre, blk32, 0, stream>>>(x0_bf, wih_bf[0], preF0, nullptr, SEQ, G4, KP, bias4 + 0 * G4, nullptr, 0);
  k_gemm_bf16_wmma<<<grid_pre, blk32, 0, stream>>>(x0_bf, wih_bf[1], preB0, nullptr, SEQ, G4, KP, bias4 + 1 * G4, nullptr, 0);
  k_lstm_bidir<<<2, G4, 0, stream>>>(preF0, preB0, wpk[0], wpk[1], x1_bf);

  // ---- LSTM layer 1 ----
  k_gemm_bf16_wmma<<<grid_pre, blk32, 0, stream>>>(x1_bf, wih_bf[2], preF1, nullptr, SEQ, G4, KP, bias4 + 2 * G4, nullptr, 0);
  k_gemm_bf16_wmma<<<grid_pre, blk32, 0, stream>>>(x1_bf, wih_bf[3], preB1, nullptr, SEQ, G4, KP, bias4 + 3 * G4, nullptr, 0);
  k_lstm_bidir<<<2, G4, 0, stream>>>(preF1, preB1, wpk[2], wpk[3], x2_bf);

  // ---- head / dep MLPs (bf16-out fused, fp32 skipped) ----
  k_gemm_bf16_wmma<<<grid_400, blk32, 0, stream>>>(x2_bf, wh1_bf, nullptr, h1_bf,   SEQ, INF_, KP, b_h1, nullptr, 1);
  k_gemm_bf16_wmma<<<grid_400, blk32, 0, stream>>>(x2_bf, wd1_bf, nullptr, d1_bf,   SEQ, INF_, KP, b_d1, nullptr, 1);
  k_gemm_bf16_wmma<<<grid_400, blk32, 0, stream>>>(h1_bf, wh2_bf, nullptr, head_bf, SEQ, INF_, KP, b_h2, nullptr, 1);
  k_gemm_bf16_wmma<<<grid_400, blk32, 0, stream>>>(d1_bf, wd2_bf, nullptr, dep_bf,  SEQ, INF_, KP, b_d2, nullptr, 1);

  // ---- bilinear: tmp = head @ W_bi ; out = tmp @ dep^T + b ----
  k_gemm_bf16_wmma<<<grid_400, blk32, 0, stream>>>(head_bf, wbiT_bf, nullptr, tmp_bf, SEQ, INF_, KP, nullptr, nullptr, 0);
  k_gemm_bf16_wmma<<<grid_out, blk32, 0, stream>>>(tmp_bf, dep_bf, out, nullptr, SEQ, SEQ, KP, nullptr, b_bi, 0);
}
